// MERGE_TAMDUR_128849019014
// MI455X (gfx1250) — matrix operand order repaired and checked by reading, not compiled, not run on hardware
//
#include <hip/hip_runtime.h>
#include <hip/hip_bf16.h>

#define NN     50000
#define FF     160
#define EE     500000
#define DKDIM  64
#define NPAD   50032   // round_up(N + S_EXTRA, 16)
#define N3     50002   // N + G_EXTRA
#define N4     50020   // N + S_EXTRA
#define NTILES 10      // FF / 16

typedef __bf16 v16bf __attribute__((ext_vector_type(16)));
typedef float  v8f   __attribute__((ext_vector_type(8)));

union BfFrag { v16bf v; unsigned int u[8]; };

// ---------- helpers ----------
static __device__ inline unsigned short f2bf(float f) {
    unsigned int u = __float_as_uint(f);
    unsigned int r = (u + 0x7FFFu + ((u >> 16) & 1u)) >> 16;   // RNE
    return (unsigned short)r;
}
static __device__ inline unsigned int fenc(float x) {          // order-preserving encoding
    unsigned int u = __float_as_uint(x);
    return (u & 0x80000000u) ? ~u : (u | 0x80000000u);
}
static __device__ inline float fdec(unsigned int e) {
    unsigned int u = (e & 0x80000000u) ? (e & 0x7FFFFFFFu) : ~e;
    return __uint_as_float(u);
}

// ---------- build x = concat(aft_rnn, static), fp32 + bf16, zero pad rows ----------
__global__ void build_x_kernel(const float* __restrict__ aft, const float* __restrict__ stat,
                               float* __restrict__ xf, unsigned short* __restrict__ xb) {
    int idx = blockIdx.x * blockDim.x + threadIdx.x;
    int total = NPAD * FF;
    if (idx >= total) return;
    int r = idx / FF, f = idx - r * FF;
    float v = 0.f;
    if (r < NN) v = (f < 144) ? aft[r * 144 + f] : stat[r * 16 + (f - 144)];
    xf[idx] = v;
    xb[idx] = f2bf(v);
}

__global__ void f2bf_kernel(const float* __restrict__ src, unsigned short* __restrict__ dst, int count) {
    int idx = blockIdx.x * blockDim.x + threadIdx.x;
    for (; idx < count; idx += gridDim.x * blockDim.x) dst[idx] = f2bf(src[idx]);
}

__global__ void zero_kernel(float* __restrict__ dst, int count) {
    int idx = blockIdx.x * blockDim.x + threadIdx.x;
    for (; idx < count; idx += gridDim.x * blockDim.x) dst[idx] = 0.f;
}

// ---------- WMMA GEMM: C[NPAD x 160] = A[NPAD x 160](bf16) @ B[160 x 160](bf16) ----------
// One wave per 16-row output stripe: A fragment loaded once per K-step and
// reused across all 10 N-tiles (10 accumulators, 50 WMMAs per wave).
__global__ void wmma_gemm160_kernel(const unsigned short* __restrict__ A,
                                    const unsigned short* __restrict__ B,
                                    float* __restrict__ C) {
    int lane = threadIdx.x;            // 0..31 (wave32)
    int m0   = blockIdx.x << 4;
    int mr   = lane & 15;
    int half = lane >> 4;
    size_t arow_off = (size_t)(m0 + mr) * FF;

    v8f acc[NTILES];
    #pragma unroll
    for (int nt = 0; nt < NTILES; ++nt) acc[nt] = (v8f){};

    #pragma unroll
    for (int k0 = 0; k0 < FF; k0 += 32) {
        // A fragment: 16-bit A 16x32 layout (pairs of K per dword; halves at K+8 / K+24)
        BfFrag a;
        const unsigned int* arow = (const unsigned int*)(A + arow_off + k0);
        #pragma unroll
        for (int v = 0; v < 8; ++v) {
            int kk = (v & 3) * 2 + ((v >> 2) << 4) + (half << 3); // bf16 index (even)
            a.u[v] = arow[kk >> 1];
        }
        // B fragments: lane = N column, K in same order as A (pairs of K per dword; halves at K+8 / K+24)
        const unsigned short* bcol = B + (size_t)k0 * FF + mr;
        #pragma unroll
        for (int nt = 0; nt < NTILES; ++nt) {
            BfFrag b;
            #pragma unroll
            for (int v = 0; v < 8; ++v) {
                int kk = (v & 3) * 2 + ((v >> 2) << 4) + (half << 3);
                b.u[v] = (unsigned int)bcol[(size_t)kk * FF + nt * 16] |
                         ((unsigned int)bcol[(size_t)(kk + 1) * FF + nt * 16] << 16);
            }
            acc[nt] = __builtin_amdgcn_wmma_f32_16x16x32_bf16(false, a.v, false, b.v,
                                                              (short)0, acc[nt], false, false);
        }
    }
    // C/D layout: N = lane&15, M = v + 8*half
    #pragma unroll
    for (int nt = 0; nt < NTILES; ++nt) {
        #pragma unroll
        for (int v = 0; v < 8; ++v) {
            int m = v + (half << 3);
            C[(size_t)(m0 + m) * FF + nt * 16 + mr] = acc[nt][v];
        }
    }
}

// ---------- el/er projections + per-node init ----------
__global__ void elr_kernel(const float* __restrict__ h, const float* __restrict__ al,
                           const float* __restrict__ ar, float* __restrict__ el,
                           float* __restrict__ er, unsigned int* __restrict__ menc,
                           float* __restrict__ ssum, int n) {
    int i = blockIdx.x * blockDim.x + threadIdx.x;
    if (i >= n) return;
    const float* row = h + (size_t)i * FF;
    float a = 0.f, b = 0.f;
    #pragma unroll 4
    for (int f = 0; f < FF; ++f) { float v = row[f]; a += v * al[f]; b += v * ar[f]; }
    el[i] = a; er[i] = b;
    menc[i] = 0u;      // below enc of any finite value
    ssum[i] = 0.f;
}

// ---------- edge phase 1: e = leaky_relu(el[src]+er[dst], 0.2); segment max ----------
__global__ void edge_e_kernel(const int* __restrict__ src, const int* __restrict__ dst,
                              const float* __restrict__ el, const float* __restrict__ er,
                              float* __restrict__ ebuf, unsigned int* __restrict__ menc) {
    int i = blockIdx.x * blockDim.x + threadIdx.x;
    if (i >= EE) return;
    float v = el[src[i]] + er[dst[i]];
    if (v < 0.f) v *= 0.2f;
    ebuf[i] = v;
    atomicMax(&menc[dst[i]], fenc(v));
}

__global__ void m_fix_kernel(const unsigned int* __restrict__ menc, float* __restrict__ mfl, int n) {
    int i = blockIdx.x * blockDim.x + threadIdx.x;
    if (i >= n) return;
    float v = fdec(menc[i]);
    mfl[i] = isfinite(v) ? v : 0.f;     // matches jnp.where(isfinite(m), m, 0)
}

// ---------- edge phase 2: ex = exp(e - m[dst]); segment sum ----------
__global__ void edge_ex_kernel(const int* __restrict__ dst, const float* __restrict__ ebuf,
                               const float* __restrict__ mfl, float* __restrict__ exbuf,
                               float* __restrict__ ssum) {
    int i = blockIdx.x * blockDim.x + threadIdx.x;
    if (i >= EE) return;
    int d = dst[i];
    float ex = __expf(ebuf[i] - mfl[d]);
    exbuf[i] = ex;
    atomicAdd(&ssum[d], ex);
}

// ---------- edge phase 3: out[dst] += alpha * h[src]  (32 lanes per edge) ----------
__global__ void edge_agg_kernel(const int* __restrict__ src, const int* __restrict__ dst,
                                const float* __restrict__ exbuf, const float* __restrict__ ssum,
                                const float* __restrict__ h, float* __restrict__ out) {
    int t = blockIdx.x * blockDim.x + threadIdx.x;
    int e = t >> 5, lane = t & 31;
    if (e >= EE) return;
    int s = src[e], d = dst[e];
    float alpha = exbuf[e] / fmaxf(ssum[d], 1e-9f);
    const float* hrow = h + (size_t)s * FF;
    float* orow = out + (size_t)d * FF;
    #pragma unroll
    for (int f = lane; f < FF; f += 32)
        atomicAdd(&orow[f], alpha * hrow[f]);
}

// ---------- epilogue: out = leaky_relu(out + b, 0.01) ----------
__global__ void bias_act_kernel(float* __restrict__ out, const float* __restrict__ b, int n) {
    int idx = blockIdx.x * blockDim.x + threadIdx.x;
    if (idx >= n * FF) return;
    int f = idx % FF;
    float v = out[idx] + b[f];
    out[idx] = (v < 0.f) ? v * 0.01f : v;
}

// ---------- Q = x @ Wq + bq ----------
__global__ void q_kernel(const float* __restrict__ xf, const float* __restrict__ Wq,
                         const float* __restrict__ bq, float* __restrict__ Q) {
    int idx = blockIdx.x * blockDim.x + threadIdx.x;
    if (idx >= NN * DKDIM) return;
    int n = idx >> 6, d = idx & 63;
    const float* xr = xf + (size_t)n * FF;
    float acc = bq[d];
    #pragma unroll 4
    for (int f = 0; f < FF; ++f) acc += xr[f] * Wq[f * DKDIM + d];
    Q[idx] = acc;
}

// ---------- attention: scores over 5 keys -> softmax alpha ----------
__global__ void attn_alpha_kernel(const float* __restrict__ Q, const float* __restrict__ Wk,
                                  const float* __restrict__ bk,
                                  const float* __restrict__ a1, const float* __restrict__ a2,
                                  const float* __restrict__ a3, const float* __restrict__ a4,
                                  const float* __restrict__ ady, float* __restrict__ alpha) {
    int n = blockIdx.x * blockDim.x + threadIdx.x;
    if (n >= NN) return;
    const float* Xs[5] = { a1 + (size_t)n * FF, a2 + (size_t)n * FF, a3 + (size_t)n * FF,
                           a4 + (size_t)n * FF, ady + (size_t)n * FF };
    const float* q = Q + (size_t)n * DKDIM;
    float sc[5], mx = -1e30f;
    for (int k = 0; k < 5; ++k) {
        const float* xr = Xs[k];
        float acc = 0.f;
        for (int d = 0; d < DKDIM; ++d) {
            float kd = bk[d];
            #pragma unroll 4
            for (int f = 0; f < FF; ++f) kd += xr[f] * Wk[f * DKDIM + d];
            acc += q[d] * kd;
        }
        sc[k] = acc * 0.125f;           // 1/sqrt(64)
        mx = fmaxf(mx, sc[k]);
    }
    float s = 0.f;
    for (int k = 0; k < 5; ++k) { float e = __expf(sc[k] - mx); alpha[n * 5 + k] = e; s += e; }
    float inv = 1.f / s;
    for (int k = 0; k < 5; ++k) alpha[n * 5 + k] *= inv;
}

// ---------- merged + final linear + log_softmax ----------
__global__ void final_kernel(const float* __restrict__ xf, const float* __restrict__ alpha,
                             const float* __restrict__ Wl, const float* __restrict__ bl,
                             const float* __restrict__ a1, const float* __restrict__ a2,
                             const float* __restrict__ a3, const float* __restrict__ a4,
                             const float* __restrict__ ady, float* __restrict__ out) {
    int n = blockIdx.x * blockDim.x + threadIdx.x;
    if (n >= NN) return;
    const float* Xs[5] = { a1 + (size_t)n * FF, a2 + (size_t)n * FF, a3 + (size_t)n * FF,
                           a4 + (size_t)n * FF, ady + (size_t)n * FF };
    float al5[5];
    for (int k = 0; k < 5; ++k) al5[k] = alpha[n * 5 + k];
    const float* xr = xf + (size_t)n * FF;
    float acc0 = bl[0], acc1 = bl[1];
    #pragma unroll 4
    for (int f = 0; f < FF; ++f) {
        float v = xr[f];
        acc0 += v * Wl[f * 2 + 0];
        acc1 += v * Wl[f * 2 + 1];
    }
    for (int f = 0; f < FF; ++f) {
        float mf = 0.f;
        for (int k = 0; k < 5; ++k) mf += al5[k] * Xs[k][f];
        acc0 += mf * Wl[(FF + f) * 2 + 0];
        acc1 += mf * Wl[(FF + f) * 2 + 1];
    }
    float mx = fmaxf(acc0, acc1);
    float lse = mx + __logf(__expf(acc0 - mx) + __expf(acc1 - mx));
    out[n * 2 + 0] = acc0 - lse;
    out[n * 2 + 1] = acc1 - lse;
}

// ============================ host orchestration ============================
extern "C" void kernel_launch(void* const* d_in, const int* in_sizes, int n_in,
                              void* d_out, int out_size, void* d_ws, size_t ws_size,
                              hipStream_t stream) {
    const float* aft    = (const float*)d_in[0];
    const float* stat   = (const float*)d_in[1];
    const int*   srcs[7] = { (const int*)d_in[2],  (const int*)d_in[4],  (const int*)d_in[6],
                             (const int*)d_in[8],  (const int*)d_in[10], (const int*)d_in[12],
                             (const int*)d_in[14] };
    const int*   dsts[7] = { (const int*)d_in[3],  (const int*)d_in[5],  (const int*)d_in[7],
                             (const int*)d_in[9],  (const int*)d_in[11], (const int*)d_in[13],
                             (const int*)d_in[15] };
    const float* gat_W  = (const float*)d_in[16];
    const float* gat_al = (const float*)d_in[17];
    const float* gat_ar = (const float*)d_in[18];
    const float* gat_b  = (const float*)d_in[19];
    const float* Wq     = (const float*)d_in[20];
    const float* bq     = (const float*)d_in[21];
    const float* Wk     = (const float*)d_in[22];
    const float* bk     = (const float*)d_in[23];
    const float* Wl     = (const float*)d_in[24];
    const float* bl     = (const float*)d_in[25];
    float* out = (float*)d_out;

    // bump allocator on workspace
    char* p = (char*)d_ws;
    auto alloc = [&](size_t bytes) -> void* {
        void* r = (void*)p;
        p += (bytes + 255) & ~(size_t)255;
        return r;
    };
    const size_t NF = (size_t)NPAD * FF;
    float*          xf    = (float*)alloc(NF * 4);
    unsigned short* xb    = (unsigned short*)alloc(NF * 2);
    unsigned short* hb    = (unsigned short*)alloc(NF * 2);
    unsigned short* Wb    = (unsigned short*)alloc((size_t)7 * FF * FF * 2);
    float*          h     = (float*)alloc(NF * 4);
    float*          el    = (float*)alloc((size_t)NPAD * 4);
    float*          er    = (float*)alloc((size_t)NPAD * 4);
    unsigned int*   menc  = (unsigned int*)alloc((size_t)NPAD * 4);
    float*          mfl   = (float*)alloc((size_t)NPAD * 4);
    float*          ssum  = (float*)alloc((size_t)NPAD * 4);
    float*          ebuf  = (float*)alloc((size_t)EE * 4);
    float*          exbuf = (float*)alloc((size_t)EE * 4);
    float*          a_dy  = (float*)alloc(NF * 4);
    float*          a1    = (float*)alloc(NF * 4);
    float*          a2    = (float*)alloc(NF * 4);
    float*          a3    = (float*)alloc(NF * 4);
    float*          a4    = (float*)alloc(NF * 4);
    float*          mid   = (float*)alloc(NF * 4);
    float*          Qbuf  = (float*)alloc((size_t)NN * DKDIM * 4);
    float*          alpha = (float*)alloc((size_t)NN * 5 * 4);

    const int TB = 256;

    // build x (fp32 + bf16), convert all 7 weight matrices to bf16
    build_x_kernel<<<(int)((NF + TB - 1) / TB), TB, 0, stream>>>(aft, stat, xf, xb);
    {
        int wcnt = 7 * FF * FF;
        f2bf_kernel<<<(wcnt + TB - 1) / TB, TB, 0, stream>>>(gat_W, Wb, wcnt);
    }

    auto run_gat = [&](int li, const int* src, const int* dst,
                       const unsigned short* inB, float* outBuf, int n) {
        wmma_gemm160_kernel<<<NPAD / 16, 32, 0, stream>>>(inB, Wb + (size_t)li * FF * FF, h);
        elr_kernel<<<(n + TB - 1) / TB, TB, 0, stream>>>(h, gat_al + li * FF, gat_ar + li * FF,
                                                         el, er, menc, ssum, n);
        zero_kernel<<<4096, TB, 0, stream>>>(outBuf, (int)NF);
        edge_e_kernel<<<(EE + TB - 1) / TB, TB, 0, stream>>>(src, dst, el, er, ebuf, menc);
        m_fix_kernel<<<(n + TB - 1) / TB, TB, 0, stream>>>(menc, mfl, n);
        edge_ex_kernel<<<(EE + TB - 1) / TB, TB, 0, stream>>>(dst, ebuf, mfl, exbuf, ssum);
        edge_agg_kernel<<<(int)(((size_t)EE * 32 + TB - 1) / TB), TB, 0, stream>>>(
            src, dst, exbuf, ssum, h, outBuf);
        bias_act_kernel<<<(n * FF + TB - 1) / TB, TB, 0, stream>>>(outBuf, gat_b + li * FF, n);
    };

    // branch GATs
    run_gat(0, srcs[0], dsts[0], xb, a_dy, NN);
    run_gat(1, srcs[1], dsts[1], xb, a1, NN);
    run_gat(2, srcs[2], dsts[2], xb, a2, NN);
    // stacked GAT over x3 (N+2 nodes)
    run_gat(3, srcs[3], dsts[3], xb, mid, N3);
    f2bf_kernel<<<(int)((NF + TB - 1) / TB), TB, 0, stream>>>(mid, hb, (int)NF);
    run_gat(4, srcs[4], dsts[4], hb, a3, N3);
    // stacked GAT over x4 (N+20 nodes)
    run_gat(5, srcs[5], dsts[5], xb, mid, N4);
    f2bf_kernel<<<(int)((NF + TB - 1) / TB), TB, 0, stream>>>(mid, hb, (int)NF);
    run_gat(6, srcs[6], dsts[6], hb, a4, N4);

    // attention + classifier
    q_kernel<<<(NN * DKDIM + TB - 1) / TB, TB, 0, stream>>>(xf, Wq, bq, Qbuf);
    attn_alpha_kernel<<<(NN + TB - 1) / TB, TB, 0, stream>>>(Qbuf, Wk, bk, a1, a2, a3, a4, a_dy, alpha);
    final_kernel<<<(NN + TB - 1) / TB, TB, 0, stream>>>(xf, alpha, Wl, bl, a1, a2, a3, a4, a_dy, out);
}
